// GridTransformerBlock_1331439861927
// MI455X (gfx1250) — matrix operand order repaired and checked by reading, not compiled, not run on hardware
//
#include <hip/hip_runtime.h>
#include <hip/hip_bf16.h>
#include <math.h>

// ---------------- problem constants ----------------
#define E_DIM   384
#define H_HEADS 8
#define D_HEAD  48
#define FF_DIM  1536
#define B_BATCH 8
#define S_SEQ   (128*128)
#define M_TOK   (B_BATCH * S_SEQ)      // 131072 tokens

typedef __attribute__((ext_vector_type(16))) __bf16 bf16x16;
typedef __attribute__((ext_vector_type(8)))  float  f32x8;

union ABfrag { bf16x16 v; unsigned u[8]; };

__device__ __forceinline__ unsigned short f2bf(float f) {
    unsigned u = __float_as_uint(f);
    unsigned r = u + 0x7FFFu + ((u >> 16) & 1u);   // round-nearest-even
    return (unsigned short)(r >> 16);
}
__device__ __forceinline__ float bf2f(unsigned short h) {
    return __uint_as_float(((unsigned)h) << 16);
}

// ---------------- fp32 -> bf16 weight conversion ----------------
__global__ __launch_bounds__(256)
void f32_to_bf16_kernel(const float* __restrict__ in, unsigned short* __restrict__ out, int n) {
    int i = blockIdx.x * 256 + threadIdx.x;
    if (i < n) out[i] = f2bf(in[i]);
}

// ---------------- LayerNorm (one wave32 per token, E=384) ----------------
__global__ __launch_bounds__(256)
void ln_bf16_kernel(const float* __restrict__ x, const float* __restrict__ gam,
                    const float* __restrict__ bet, unsigned short* __restrict__ out) {
    const int lane = threadIdx.x & 31;
    const int wv   = threadIdx.x >> 5;
    const int row  = blockIdx.x * 8 + wv;
    const float* xr = x + (size_t)row * E_DIM;

    float vals[12];
    float s = 0.f;
#pragma unroll
    for (int i = 0; i < 12; i++) { vals[i] = xr[lane + 32 * i]; s += vals[i]; }
#pragma unroll
    for (int off = 16; off; off >>= 1) s += __shfl_xor(s, off, 32);
    float mean = s * (1.0f / E_DIM);

    float vs = 0.f;
#pragma unroll
    for (int i = 0; i < 12; i++) { float d = vals[i] - mean; vs += d * d; }
#pragma unroll
    for (int off = 16; off; off >>= 1) vs += __shfl_xor(vs, off, 32);
    float rstd = rsqrtf(vs * (1.0f / E_DIM) + 1e-5f);

    unsigned short* orow = out + (size_t)row * E_DIM;
#pragma unroll
    for (int i = 0; i < 12; i++) {
        int e = lane + 32 * i;
        orow[e] = f2bf((vals[i] - mean) * rstd * gam[e] + bet[e]);
    }
}

// ---------------- bf16 WMMA GEMM: Y = epi(X[M,K] @ W[N,K]^T + bias) ----------------
// blockIdx.x = n-strip (fast-varying in dispatch order -> blocks sharing the same
// X rows run concurrently and hit L2), blockIdx.y = 128-row m-block.
// Each wave computes a 16 x (16*NQ) strip; all NQ B fragments are staged in
// registers before the WMMA chain so loads retire against counted waits.
// EPI 0: store bf16     EPI 1: exact-erf GELU, store bf16     EPI 2: +resid, store fp32
template <int EPI, int NQ>
__global__ __launch_bounds__(256)
void gemm_bf16_kernel(const unsigned short* __restrict__ X,
                      const unsigned short* __restrict__ W,
                      const float* __restrict__ bias,
                      const float* __restrict__ resid,
                      void* __restrict__ outp,
                      int N, int K) {
    const int lane  = threadIdx.x & 31;
    const int wv    = threadIdx.x >> 5;
    const int lo16  = lane & 15;
    const int hi    = lane >> 4;
    const int mBase = blockIdx.y * 128 + wv * 16;
    const int nBase = blockIdx.x * (16 * NQ);

    f32x8 acc[NQ];
#pragma unroll
    for (int j = 0; j < NQ; j++) acc[j] = (f32x8){0.f,0.f,0.f,0.f,0.f,0.f,0.f,0.f};

    const unsigned short* xrow = X + (size_t)(mBase + lo16) * K;
    const unsigned short* wrow0 = W + (size_t)lo16 * K + hi * 8;

    for (int kk = 0; kk < K; kk += 32) {
        // prefetch next K-chunk of the activation row (global_prefetch_b8)
        if (kk + 32 < K) __builtin_prefetch(xrow + kk + 32 + hi * 8, 0, 0);

        ABfrag a;
#pragma unroll
        for (int v = 0; v < 8; v++) {
            // 16-bit A 16x32 layout: K = kk + (v>=4?16:0) + (lane/16)*8 + (v%4)*2
            int k = kk + ((v & 4) ? 16 : 0) + hi * 8 + (v & 3) * 2;
            a.u[v] = *(const unsigned*)(xrow + k);
        }
        // stage all NQ B fragments first (one big load clause)
        ABfrag b[NQ];
#pragma unroll
        for (int j = 0; j < NQ; j++) {
            // 16-bit B 32x16 layout: N = lane%16, K = kk + (v>=4?16:0) + (lane/16)*8 + (v%4)*2
            const unsigned short* wrow = wrow0 + (size_t)(nBase + j * 16) * K + kk;
#pragma unroll
            for (int v = 0; v < 8; v++) b[j].u[v] = *(const unsigned*)(wrow + ((v & 4) ? 16 : 0) + (v & 3) * 2);
        }
#pragma unroll
        for (int j = 0; j < NQ; j++) {
            acc[j] = __builtin_amdgcn_wmma_f32_16x16x32_bf16(
                false, a.v, false, b[j].v, (short)0, acc[j], false, false);
        }
    }

#pragma unroll
    for (int j = 0; j < NQ; j++) {
        const int n  = nBase + j * 16 + lo16;
        const float bn = bias[n];
#pragma unroll
        for (int v = 0; v < 8; v++) {
            const int m = mBase + hi * 8 + v;     // C/D layout: M = vgpr + (lane/16)*8
            float val = acc[j][v] + bn;
            size_t oi = (size_t)m * N + n;
            if (EPI == 0) {
                ((unsigned short*)outp)[oi] = f2bf(val);
            } else if (EPI == 1) {
                float g = 0.5f * val * (1.0f + erff(val * 0.70710678118654752f));
                ((unsigned short*)outp)[oi] = f2bf(g);
            } else {
                ((float*)outp)[oi] = val + resid[oi];
            }
        }
    }
}

// ---------------- batch-axis attention (one thread per (window,t,head,b)) ----------------
// qkv rows: [tok][1152] = q|k|v each E=384; 8x8 softmax across the batch axis.
__global__ __launch_bounds__(256)
void grid_attn_kernel(const unsigned short* __restrict__ qkv,
                      unsigned short* __restrict__ aout) {
    const int idx = blockIdx.x * 256 + threadIdx.x;   // 64*256*8*8 = 1048576
    const int b = idx & 7;
    const int h = (idx >> 3) & 7;
    const int t = (idx >> 6) & 255;
    const int w = idx >> 14;
    const int wi = w >> 3, wj = w & 7;
    const int ti = t >> 4, tj = t & 15;
    const int s = (wi * 16 + ti) * 128 + (wj * 16 + tj);

    int tok[8];
#pragma unroll
    for (int c = 0; c < 8; c++) tok[c] = c * S_SEQ + s;

    const unsigned short* qp = qkv + (size_t)tok[b] * (3 * E_DIM) + h * D_HEAD;
    float q[D_HEAD];
#pragma unroll
    for (int d = 0; d < D_HEAD; d++) q[d] = bf2f(qp[d]);

    float sc[8];
    float mx = -1e30f;
    for (int c = 0; c < 8; c++) {
        const unsigned short* kp = qkv + (size_t)tok[c] * (3 * E_DIM) + E_DIM + h * D_HEAD;
        float acc = 0.f;
#pragma unroll
        for (int d = 0; d < D_HEAD; d++) acc += q[d] * bf2f(kp[d]);
        acc *= 0.14433756729740645f;            // 1/sqrt(48)
        sc[c] = acc;
        mx = fmaxf(mx, acc);
    }
    float ssum = 0.f;
#pragma unroll
    for (int c = 0; c < 8; c++) { sc[c] = __expf(sc[c] - mx); ssum += sc[c]; }
    const float inv = 1.0f / ssum;

    float o[D_HEAD];
#pragma unroll
    for (int d = 0; d < D_HEAD; d++) o[d] = 0.f;
    for (int c = 0; c < 8; c++) {
        const unsigned short* vp = qkv + (size_t)tok[c] * (3 * E_DIM) + 2 * E_DIM + h * D_HEAD;
        const float p = sc[c] * inv;
#pragma unroll
        for (int d = 0; d < D_HEAD; d++) o[d] += p * bf2f(vp[d]);
    }
    unsigned short* op = aout + (size_t)tok[b] * E_DIM + h * D_HEAD;
#pragma unroll
    for (int d = 0; d < D_HEAD; d++) op[d] = f2bf(o[d]);
}

// ---------------- host orchestration ----------------
extern "C" void kernel_launch(void* const* d_in, const int* in_sizes, int n_in,
                              void* d_out, int out_size, void* d_ws, size_t ws_size,
                              hipStream_t stream) {
    (void)in_sizes; (void)n_in; (void)out_size; (void)ws_size;
    const float* x      = (const float*)d_in[0];
    const float* wqkv_f = (const float*)d_in[1];
    const float* bqkv   = (const float*)d_in[2];
    const float* wout_f = (const float*)d_in[3];
    const float* bout   = (const float*)d_in[4];
    const float* ln1g   = (const float*)d_in[5];
    const float* ln1b   = (const float*)d_in[6];
    const float* ln2g   = (const float*)d_in[7];
    const float* ln2b   = (const float*)d_in[8];
    const float* w1_f   = (const float*)d_in[9];
    const float* b1     = (const float*)d_in[10];
    const float* w2_f   = (const float*)d_in[11];
    const float* b2     = (const float*)d_in[12];
    float* out = (float*)d_out;

    // workspace layout (all offsets 256B aligned)
    char* ws = (char*)d_ws;
    const size_t XN_BYTES   = (size_t)M_TOK * E_DIM * 2;          // 100,663,296
    const size_t QKV_BYTES  = (size_t)M_TOK * 3 * E_DIM * 2;      // 301,989,888
    const size_t AOUT_BYTES = (size_t)M_TOK * E_DIM * 2;          // 100,663,296
    unsigned short* xn   = (unsigned short*)(ws);                          // LN output (bf16), reused for LN2
    unsigned short* qkv  = (unsigned short*)(ws + XN_BYTES);               // qkv bf16
    unsigned short* aout = (unsigned short*)(ws + XN_BYTES + QKV_BYTES);   // attn out bf16
    unsigned short* gbuf = (unsigned short*)(ws + XN_BYTES);               // GELU buf aliases qkv+aout (same size)
    char* wsec = ws + XN_BYTES + QKV_BYTES + AOUT_BYTES;
    unsigned short* wqkv = (unsigned short*)(wsec);
    unsigned short* wout = (unsigned short*)(wsec + 884736);
    unsigned short* w1b  = (unsigned short*)(wsec + 884736 + 294912);
    unsigned short* w2b  = (unsigned short*)(wsec + 884736 + 294912 + 1179648);

    // 1) weights -> bf16
    f32_to_bf16_kernel<<<(3*E_DIM*E_DIM + 255)/256, 256, 0, stream>>>(wqkv_f, wqkv, 3*E_DIM*E_DIM);
    f32_to_bf16_kernel<<<(E_DIM*E_DIM   + 255)/256, 256, 0, stream>>>(wout_f, wout, E_DIM*E_DIM);
    f32_to_bf16_kernel<<<(FF_DIM*E_DIM  + 255)/256, 256, 0, stream>>>(w1_f,  w1b,  FF_DIM*E_DIM);
    f32_to_bf16_kernel<<<(E_DIM*FF_DIM  + 255)/256, 256, 0, stream>>>(w2_f,  w2b,  E_DIM*FF_DIM);

    // 2) LN1 -> xn (bf16)
    ln_bf16_kernel<<<M_TOK/8, 256, 0, stream>>>(x, ln1g, ln1b, xn);

    // 3) QKV GEMM: [M,384] x [1152,384]^T -> qkv bf16   (12 n-strips of 96, n fast)
    gemm_bf16_kernel<0,6><<<dim3((3*E_DIM)/96, M_TOK/128), 256, 0, stream>>>(
        xn, wqkv, bqkv, nullptr, qkv, 3*E_DIM, E_DIM);

    // 4) batch-axis windowed attention -> aout bf16
    grid_attn_kernel<<<(64*256*8*8)/256, 256, 0, stream>>>(qkv, aout);

    // 5) out-proj GEMM + residual x -> d_out (fp32)   (4 n-strips share X rows via L2)
    gemm_bf16_kernel<2,6><<<dim3(E_DIM/96, M_TOK/128), 256, 0, stream>>>(
        aout, wout, bout, x, out, E_DIM, E_DIM);

    // 6) LN2(d_out) -> xn (bf16)
    ln_bf16_kernel<<<M_TOK/8, 256, 0, stream>>>(out, ln2g, ln2b, xn);

    // 7) MLP1 GEMM + exact-erf GELU -> gbuf bf16 [M,1536]   (16 n-strips of 96)
    gemm_bf16_kernel<1,6><<<dim3(FF_DIM/96, M_TOK/128), 256, 0, stream>>>(
        xn, w1b, b1, nullptr, gbuf, FF_DIM, E_DIM);

    // 8) MLP2 GEMM + residual d_out -> d_out (fp32)
    //    n-strip in blockIdx.x: the 4 blocks reading the same 128 gbuf rows are
    //    dispatched adjacently, so the 402MB gbuf streams from HBM ~once via L2.
    gemm_bf16_kernel<2,6><<<dim3(E_DIM/96, M_TOK/128), 256, 0, stream>>>(
        gbuf, w2b, b2, out, out, E_DIM, FF_DIM);
}
